// Attention_35639638622507
// MI455X (gfx1250) — matrix operand order repaired and checked by reading, not compiled, not run on hardware
//
#include <hip/hip_runtime.h>
#include <hip/hip_bf16.h>

// ---------------------------------------------------------------------------
// Types for CDNA5 WMMA (wave32): D(16x16 f32) = A(16x32 bf16) x B(32x16 bf16) + C
// ---------------------------------------------------------------------------
typedef __bf16 bf16_t;
typedef __attribute__((ext_vector_type(16))) __bf16 v16bf;
typedef __attribute__((ext_vector_type(8)))  __bf16 v8bf;
typedef __attribute__((ext_vector_type(4)))  __bf16 v4bf;
typedef __attribute__((ext_vector_type(8)))  float  v8f;
typedef __attribute__((ext_vector_type(4)))  int    v4i;

__device__ __forceinline__ v8f wmma_bf16(v16bf a, v16bf b, v8f c) {
  return __builtin_amdgcn_wmma_f32_16x16x32_bf16(false, a, false, b, (short)0, c,
                                                 false, false);
}

__device__ __forceinline__ v16bf cat8(v8bf a, v8bf b) {
  return __builtin_shufflevector(a, b, 0, 1, 2, 3, 4, 5, 6, 7, 8, 9, 10, 11, 12, 13, 14, 15);
}

// CDNA5 async global->LDS staging (guarded: falls back to direct loads)
#if defined(__has_builtin)
#if __has_builtin(__builtin_amdgcn_global_load_async_to_lds_b128) && \
    __has_builtin(__builtin_amdgcn_s_wait_asynccnt)
#define HAVE_ASYNC_LDS 1
#endif
#endif
#ifndef HAVE_ASYNC_LDS
#define HAVE_ASYNC_LDS 0
#endif

#if HAVE_ASYNC_LDS
typedef __attribute__((address_space(1))) v4i* gptr_v4i;   // global v4i*
typedef __attribute__((address_space(3))) v4i* lptr_v4i;   // LDS v4i*

__device__ __forceinline__ void async_cp16(const __bf16* g, __bf16* l) {
  __bf16* gm = const_cast<__bf16*>(g);
  __builtin_amdgcn_global_load_async_to_lds_b128((gptr_v4i)gm, (lptr_v4i)l, 0, 0);
}
#endif

// Problem constants
#define BATCH 32
#define SEQ   512
#define DIM   512
#define NHEAD 8
#define DHEAD 64
#define QKVW  1536           // 3*NHEAD*DHEAD
#define RELROWS 1025
#define SCALE_Q 0.125f       // 64^-0.5

// ---------------------------------------------------------------------------
// Prep kernels
// ---------------------------------------------------------------------------
__global__ void cvt_f32_bf16(const float* __restrict__ src, bf16_t* __restrict__ dst, int n) {
  int i = blockIdx.x * blockDim.x + threadIdx.x;
  if (i < n) dst[i] = (bf16_t)src[i];
}

// src[rows][cols] (f32) -> dst[cols][rows] (bf16)
__global__ void transpose_cvt(const float* __restrict__ src, bf16_t* __restrict__ dst,
                              int rows, int cols) {
  int c = blockIdx.x * 32 + (threadIdx.x & 31);
  int r = blockIdx.y * 8 + (threadIdx.x >> 5);
  if (c < cols && r < rows) dst[(size_t)c * rows + r] = (bf16_t)src[(size_t)r * cols + c];
}

// ---------------------------------------------------------------------------
// Kernel 1: qkv = x @ W_qkv (M=16384, K=512, N=1536). 32x64 tile per wave,
// double-buffered fragments (8 WMMA per k-step). Scatters Qh/Kh [b,h,n,d]
// (Q pre-scaled by d^-1/2) and V transposed Vt [b,h,d,n].
// ---------------------------------------------------------------------------
__global__ __launch_bounds__(256) void qkv_gemm(const bf16_t* __restrict__ xbf,
                                                const bf16_t* __restrict__ wt,
                                                bf16_t* __restrict__ Qh,
                                                bf16_t* __restrict__ Kh,
                                                bf16_t* __restrict__ Vt) {
  const int lane = threadIdx.x & 31;
  const int wave = threadIdx.x >> 5;
  const int tile = blockIdx.x * 8 + wave;        // 512 * 24 tiles
  const int mt = tile & 511;
  const int nt = tile >> 9;                      // 0..23
  const int half = lane >> 4;
  const int l15 = lane & 15;
  const int m0 = mt * 32;

  const bf16_t* arow0 = xbf + (size_t)(m0 + l15) * DIM;
  const bf16_t* arow1 = arow0 + (size_t)16 * DIM;
  const bf16_t* bcol = wt + (size_t)(nt * 64 + l15) * DIM + half * 16;

  auto ldA = [&](const bf16_t* r, int kb) {
    return *(const v16bf*)(r + kb + half * 16);
  };
  auto ldB = [&](int ntl, int kb) {
    return *(const v16bf*)(bcol + (size_t)ntl * 16 * DIM + kb);
  };

  v8f acc[2][4] = {};
  v16bf A0 = ldA(arow0, 0), A1 = ldA(arow1, 0);
  v16bf B0 = ldB(0, 0), B1 = ldB(1, 0), B2 = ldB(2, 0), B3 = ldB(3, 0);
#pragma unroll
  for (int ks = 0; ks < 16; ++ks) {
    v16bf nA0{}, nA1{}, nB0{}, nB1{}, nB2{}, nB3{};
    if (ks < 15) {
      const int kb = (ks + 1) * 32;
      nA0 = ldA(arow0, kb); nA1 = ldA(arow1, kb);
      nB0 = ldB(0, kb); nB1 = ldB(1, kb); nB2 = ldB(2, kb); nB3 = ldB(3, kb);
    }
    acc[0][0] = wmma_bf16(A0, B0, acc[0][0]);
    acc[1][0] = wmma_bf16(A1, B0, acc[1][0]);
    acc[0][1] = wmma_bf16(A0, B1, acc[0][1]);
    acc[1][1] = wmma_bf16(A1, B1, acc[1][1]);
    acc[0][2] = wmma_bf16(A0, B2, acc[0][2]);
    acc[1][2] = wmma_bf16(A1, B2, acc[1][2]);
    acc[0][3] = wmma_bf16(A0, B3, acc[0][3]);
    acc[1][3] = wmma_bf16(A1, B3, acc[1][3]);
    A0 = nA0; A1 = nA1; B0 = nB0; B1 = nB1; B2 = nB2; B3 = nB3;
  }

  const int b = m0 >> 9;
  const int head = nt & 7;
  const int sect = nt >> 3;         // 0=Q 1=K 2=V
#pragma unroll
  for (int msub = 0; msub < 2; ++msub) {
    const int row0 = (m0 & 511) + msub * 16;
#pragma unroll
    for (int ntl = 0; ntl < 4; ++ntl) {
      const int d = ntl * 16 + l15;
      if (sect == 2) {
        bf16_t vals[8];
#pragma unroll
        for (int i = 0; i < 8; ++i) vals[i] = (bf16_t)acc[msub][ntl][i];
        const int nbase = row0 + half * 8;
        *(v8bf*)(Vt + ((size_t)((b * NHEAD + head) * DHEAD + d)) * SEQ + nbase) =
            *(const v8bf*)vals;
      } else {
        bf16_t* dst = (sect == 0) ? Qh : Kh;
        const float sc = (sect == 0) ? SCALE_Q : 1.0f;
        const size_t base = (size_t)(b * NHEAD + head) * SEQ * DHEAD;
#pragma unroll
        for (int i = 0; i < 8; ++i) {
          const int n = row0 + i + 8 * half;
          dst[base + (size_t)n * DHEAD + d] = (bf16_t)(acc[msub][ntl][i] * sc);
        }
      }
    }
  }
}

// ---------------------------------------------------------------------------
// Kernel 2: fused attention, one wave per (b, h, 16-row qtile).
// K-tiles and V-tiles staged into double-buffered LDS via async global->LDS
// (s_wait_asynccnt pipelining) when available; rel-bias band via WMMA + the
// ds_bpermute diagonal gather; two-pass softmax over the 16x512 LDS band.
// ---------------------------------------------------------------------------
#define SPITCH 516    // f32 elements per S row
#define PPITCH 528    // bf16 elements per P row

__global__ __launch_bounds__(32) void attn_kernel(const bf16_t* __restrict__ Qh,
                                                  const bf16_t* __restrict__ Kh,
                                                  const bf16_t* __restrict__ Vt,
                                                  const bf16_t* __restrict__ relbf,
                                                  bf16_t* __restrict__ AO) {
  __shared__ __align__(16) float  Sband[16 * SPITCH];
  __shared__ __align__(16) bf16_t Pband[16 * PPITCH];
  __shared__ float linv[16];
#if HAVE_ASYNC_LDS
  __shared__ __align__(16) bf16_t Ktile[2][16 * DHEAD];   // 2 x 2KB
  __shared__ __align__(16) bf16_t Vtile[2][DHEAD * 32];   // 2 x 4KB
#endif

  const int lane = threadIdx.x;
  const int half = lane >> 4;
  const int l15 = lane & 15;
  const int qt = blockIdx.x & 31;
  const int h = (blockIdx.x >> 5) & 7;
  const int b = blockIdx.x >> 8;
  const int n0 = qt * 16;

  const bf16_t* Qb = Qh + (size_t)(b * NHEAD + h) * SEQ * DHEAD;
  const bf16_t* Kb = Kh + (size_t)(b * NHEAD + h) * SEQ * DHEAD;
  const bf16_t* Vb = Vt + (size_t)(b * NHEAD + h) * DHEAD * SEQ;

#if HAVE_ASYNC_LDS
  auto stageK = [&](int jt, int buf) {
#pragma unroll
    for (int c = 0; c < 4; ++c) {
      const int eo = (c * 32 + lane) * 8;       // 16B chunks over 16x64 tile
      async_cp16(Kb + (size_t)jt * 16 * DHEAD + eo, &Ktile[buf][eo]);
    }
  };
  auto stageV = [&](int kt, int buf) {
#pragma unroll
    for (int c = 0; c < 8; ++c) {
      const int e = c * 32 + lane;              // 16B chunk id over 64x32 tile
      const int d = e >> 1;
      const int off = (e & 1) * 16;
      async_cp16(Vb + (size_t)d * SEQ + kt * 32 + off, &Vtile[buf][d * 32 + off]);
    }
  };
#endif

  // Q A-fragments for the two K-steps over d=64 (Q pre-scaled by d^-1/2)
  v16bf AQ[2];
  {
    const bf16_t* qrow = Qb + (size_t)(n0 + l15) * DHEAD;
#pragma unroll
    for (int ks = 0; ks < 2; ++ks) {
      const int ab = ks * 32 + half * 16;
      AQ[ks] = *(const v16bf*)(qrow + ab);
    }
  }

  auto ldRel = [&](int mrow, int ks) {
    return *(const v16bf*)(relbf + (size_t)mrow * DHEAD + ks * 32 + half * 16);
  };

  // ---- Pass 1: scores + relative bias into LDS ----
#if HAVE_ASYNC_LDS
  stageK(0, 0);
#endif
  // rel fragments for jt=0, register double-buffered: R[ks*2 + (c>=16)]
  v16bf R[4];
  {
    const int mb = n0 + 497;
#pragma unroll
    for (int ks = 0; ks < 2; ++ks) {
      R[ks * 2 + 0] = ldRel(mb + l15, ks);
      R[ks * 2 + 1] = ldRel(mb + 16 + l15, ks);
    }
  }
  for (int jt = 0; jt < 32; ++jt) {
    const int j0 = jt * 16;
#if HAVE_ASYNC_LDS
    if (jt < 31) { stageK(jt + 1, (jt + 1) & 1); __builtin_amdgcn_s_wait_asynccnt(4); }
    else         { __builtin_amdgcn_s_wait_asynccnt(0); }
#endif
    v16bf nR[4] = {};
    if (jt < 31) {
      const int mb = n0 - j0 - 16 + 497;
#pragma unroll
      for (int ks = 0; ks < 2; ++ks) {
        nR[ks * 2 + 0] = ldRel(mb + l15, ks);
        nR[ks * 2 + 1] = ldRel(mb + 16 + l15, ks);
      }
    }

    v8f S = v8f{};
    v8f G0 = v8f{}, G1 = v8f{};
#pragma unroll
    for (int ks = 0; ks < 2; ++ks) {
#if HAVE_ASYNC_LDS
      v16bf BK = *(const v16bf*)(&Ktile[jt & 1][l15 * DHEAD + ks * 32 + half * 16]);
#else
      v16bf BK = *(const v16bf*)(Kb + (size_t)(j0 + l15) * DHEAD + ks * 32 + half * 16);
#endif
      S = wmma_bf16(AQ[ks], BK, S);
      G0 = wmma_bf16(AQ[ks], R[ks * 2 + 0], G0);
      G1 = wmma_bf16(AQ[ks], R[ks * 2 + 1], G1);
    }
#pragma unroll
    for (int i = 0; i < 8; ++i) {
      const int c = i + 8 * half - l15 + 15;           // 0..30
      const int srcLane = (c & 15) + (half << 4);
      int g0 = __builtin_amdgcn_ds_bpermute(srcLane << 2, __builtin_bit_cast(int, G0[i]));
      int g1 = __builtin_amdgcn_ds_bpermute(srcLane << 2, __builtin_bit_cast(int, G1[i]));
      float g = __builtin_bit_cast(float, (c >= 16) ? g1 : g0);
      Sband[(i + 8 * half) * SPITCH + j0 + l15] = S[i] + g;
    }
    R[0] = nR[0]; R[1] = nR[1]; R[2] = nR[2]; R[3] = nR[3];
  }

  // ---- Pass 2: softmax stats (single wave: LDS ops in-order) ----
  {
    const int srow = l15;
    const float4* sp = (const float4*)(Sband + srow * SPITCH + half * 256);
    float mx = -1e30f;
    for (int t = 0; t < 64; ++t) {
      float4 v = sp[t];
      mx = fmaxf(mx, fmaxf(fmaxf(v.x, v.y), fmaxf(v.z, v.w)));
    }
    mx = fmaxf(mx, __shfl_xor(mx, 16));
    float sum = 0.f;
    bf16_t* pp = Pband + srow * PPITCH + half * 256;
    for (int t = 0; t < 64; ++t) {
      float4 v = sp[t];
      float e0 = __expf(v.x - mx), e1 = __expf(v.y - mx);
      float e2 = __expf(v.z - mx), e3 = __expf(v.w - mx);
      sum += (e0 + e1) + (e2 + e3);
      v4bf pk; pk.x = (bf16_t)e0; pk.y = (bf16_t)e1; pk.z = (bf16_t)e2; pk.w = (bf16_t)e3;
      *(v4bf*)(pp + 4 * t) = pk;
    }
    sum += __shfl_xor(sum, 16);
    if (half == 0) linv[l15] = 1.0f / sum;
  }

  // ---- Pass 3: O = P @ V ----
  v8f O[4] = {};
  const bf16_t* prow = Pband + l15 * PPITCH;
#if HAVE_ASYNC_LDS
  stageV(0, 0);
#endif
  for (int kt = 0; kt < 16; ++kt) {
    const int jb = kt * 32 + half * 16;
#if HAVE_ASYNC_LDS
    if (kt < 15) { stageV(kt + 1, (kt + 1) & 1); __builtin_amdgcn_s_wait_asynccnt(8); }
    else         { __builtin_amdgcn_s_wait_asynccnt(0); }
#endif
    v16bf AP = *(const v16bf*)(prow + jb);
#pragma unroll
    for (int ntl = 0; ntl < 4; ++ntl) {
      const int d = ntl * 16 + l15;
#if HAVE_ASYNC_LDS
      v16bf BV = *(const v16bf*)(&Vtile[kt & 1][d * 32 + half * 16]);
#else
      v16bf BV = *(const v16bf*)(Vb + (size_t)d * SEQ + kt * 32 + half * 16);
#endif
      O[ntl] = wmma_bf16(AP, BV, O[ntl]);
    }
  }

  float li[8];
#pragma unroll
  for (int i = 0; i < 8; ++i) li[i] = linv[i + 8 * half];
  bf16_t* aorow = AO + ((size_t)(b * SEQ + n0)) * DIM + h * DHEAD;
#pragma unroll
  for (int ntl = 0; ntl < 4; ++ntl) {
    const int d = ntl * 16 + l15;
#pragma unroll
    for (int i = 0; i < 8; ++i) {
      aorow[(size_t)(i + 8 * half) * DIM + d] = (bf16_t)(O[ntl][i] * li[i]);
    }
  }
}

// ---------------------------------------------------------------------------
// Kernel 3: out = AO @ W_out + b_out (M=16384, K=512, N=512). 32x64 wave tile.
// ---------------------------------------------------------------------------
__global__ __launch_bounds__(256) void out_gemm(const bf16_t* __restrict__ AO,
                                                const bf16_t* __restrict__ wtout,
                                                const float* __restrict__ bout,
                                                float* __restrict__ out) {
  const int lane = threadIdx.x & 31;
  const int wave = threadIdx.x >> 5;
  const int tile = blockIdx.x * 8 + wave;   // 512 * 8 tiles
  const int mt = tile & 511;
  const int nt = tile >> 9;                 // 0..7
  const int half = lane >> 4;
  const int l15 = lane & 15;
  const int m0 = mt * 32;

  const bf16_t* arow0 = AO + (size_t)(m0 + l15) * DIM;
  const bf16_t* arow1 = arow0 + (size_t)16 * DIM;
  const bf16_t* bcol = wtout + (size_t)(nt * 64 + l15) * DIM + half * 16;

  auto ldA = [&](const bf16_t* r, int kb) {
    return *(const v16bf*)(r + kb + half * 16);
  };
  auto ldB = [&](int ntl, int kb) {
    return *(const v16bf*)(bcol + (size_t)ntl * 16 * DIM + kb);
  };

  v8f acc[2][4] = {};
  v16bf A0 = ldA(arow0, 0), A1 = ldA(arow1, 0);
  v16bf B0 = ldB(0, 0), B1 = ldB(1, 0), B2 = ldB(2, 0), B3 = ldB(3, 0);
#pragma unroll
  for (int ks = 0; ks < 16; ++ks) {
    v16bf nA0{}, nA1{}, nB0{}, nB1{}, nB2{}, nB3{};
    if (ks < 15) {
      const int kb = (ks + 1) * 32;
      nA0 = ldA(arow0, kb); nA1 = ldA(arow1, kb);
      nB0 = ldB(0, kb); nB1 = ldB(1, kb); nB2 = ldB(2, kb); nB3 = ldB(3, kb);
    }
    acc[0][0] = wmma_bf16(A0, B0, acc[0][0]);
    acc[1][0] = wmma_bf16(A1, B0, acc[1][0]);
    acc[0][1] = wmma_bf16(A0, B1, acc[0][1]);
    acc[1][1] = wmma_bf16(A1, B1, acc[1][1]);
    acc[0][2] = wmma_bf16(A0, B2, acc[0][2]);
    acc[1][2] = wmma_bf16(A1, B2, acc[1][2]);
    acc[0][3] = wmma_bf16(A0, B3, acc[0][3]);
    acc[1][3] = wmma_bf16(A1, B3, acc[1][3]);
    A0 = nA0; A1 = nA1; B0 = nB0; B1 = nB1; B2 = nB2; B3 = nB3;
  }
#pragma unroll
  for (int msub = 0; msub < 2; ++msub) {
#pragma unroll
    for (int ntl = 0; ntl < 4; ++ntl) {
      const int c = nt * 64 + ntl * 16 + l15;
      const float bias = bout[c];
#pragma unroll
      for (int i = 0; i < 8; ++i) {
        out[(size_t)(m0 + msub * 16 + i + 8 * half) * DIM + c] = acc[msub][ntl][i] + bias;
      }
    }
  }
}

// ---------------------------------------------------------------------------
// Host launch
// ---------------------------------------------------------------------------
extern "C" void kernel_launch(void* const* d_in, const int* in_sizes, int n_in,
                              void* d_out, int out_size, void* d_ws, size_t ws_size,
                              hipStream_t stream) {
  const float* x    = (const float*)d_in[0];
  const float* Wqkv = (const float*)d_in[1];
  const float* rel  = (const float*)d_in[2];
  const float* Wout = (const float*)d_in[3];
  const float* bout = (const float*)d_in[4];
  float* out = (float*)d_out;

  char* ws = (char*)d_ws;
  size_t off = 0;
  auto walloc = [&](size_t bytes) -> void* {
    void* p = ws + off;
    off += (bytes + 255) & ~(size_t)255;
    return p;
  };
  const size_t M = (size_t)BATCH * SEQ;                     // 16384
  bf16_t* xbf   = (bf16_t*)walloc(M * DIM * 2);             // reused as AO
  bf16_t* wtqkv = (bf16_t*)walloc((size_t)QKVW * DIM * 2);
  bf16_t* relbf = (bf16_t*)walloc((size_t)RELROWS * DHEAD * 2);
  bf16_t* wtout = (bf16_t*)walloc((size_t)DIM * DIM * 2);
  bf16_t* Qh    = (bf16_t*)walloc(M * DHEAD * 2);           // [b,h,n,d]
  bf16_t* Kh    = (bf16_t*)walloc(M * DHEAD * 2);           // [b,h,n,d]
  bf16_t* Vt    = (bf16_t*)walloc(M * DHEAD * 2);           // [b,h,d,n]
  bf16_t* AO    = xbf;                                      // xbf dead after qkv_gemm

  cvt_f32_bf16<<<(int)(M * DIM / 256), 256, 0, stream>>>(x, xbf, (int)(M * DIM));
  transpose_cvt<<<dim3(QKVW / 32, DIM / 8), 256, 0, stream>>>(Wqkv, wtqkv, DIM, QKVW);
  cvt_f32_bf16<<<(RELROWS * DHEAD + 255) / 256, 256, 0, stream>>>(rel, relbf, RELROWS * DHEAD);
  transpose_cvt<<<dim3(DIM / 32, DIM / 8), 256, 0, stream>>>(Wout, wtout, DIM, DIM);

  // QKV projection: 512 m-tiles * 24 n-tiles / 8 waves per block
  qkv_gemm<<<1536, 256, 0, stream>>>(xbf, wtqkv, Qh, Kh, Vt);

  // Attention: one wave per (b, h, 16-row qtile)
  attn_kernel<<<BATCH * NHEAD * (SEQ / 16), 32, 0, stream>>>(Qh, Kh, Vt, relbf, AO);

  // Output projection: 512 m-tiles * 8 n-tiles / 8 waves per block
  out_gemm<<<512, 256, 0, stream>>>(AO, wtout, bout, out);
}
